// TWIN_49143015801312
// MI455X (gfx1250) — matrix operand order repaired and checked by reading, not compiled, not run on hardware
//
#include <hip/hip_runtime.h>
#include <hip/hip_bf16.h>

// ---------------------------------------------------------------------------
// Types for CDNA5 WMMA (wave32).
// ---------------------------------------------------------------------------
typedef __attribute__((ext_vector_type(16))) __bf16    v16bf;
typedef __attribute__((ext_vector_type(8)))  float     v8f;
typedef __attribute__((ext_vector_type(8)))  unsigned short v8us;

#define N_TOK   4096     // B*S
#define DMODEL  256
#define KTOP    16
#define NCODES  20

static __device__ __forceinline__ unsigned short f2bf(float f) {
    unsigned int u = __float_as_uint(f);
    unsigned int r = u + 0x7FFFu + ((u >> 16) & 1u);   // round-to-nearest-even
    return (unsigned short)(r >> 16);
}

// Build a v16bf WMMA operand from two contiguous 8-element (16B) chunks.
static __device__ __forceinline__ v16bf load16bf(const unsigned short* p0,
                                                 const unsigned short* p1) {
    union { v8us u[2]; v16bf v; } x;
    x.u[0] = *(const v8us*)p0;
    x.u[1] = *(const v8us*)p1;
    return x.v;
}

static __device__ __forceinline__ v8f wmma_bf16(v16bf a, v16bf b, v8f c) {
    return __builtin_amdgcn_wmma_f32_16x16x32_bf16(
        /*neg_a=*/false, a, /*neg_b=*/false, b,
        /*c_mod=*/(short)0, c, /*reuse_a=*/false, /*reuse_b=*/false);
}

// ---------------------------------------------------------------------------
// K0: weight prep — transpose f32 (K=256 x N) row-major -> bf16 col-major
//     [Npad][256], zero-padding columns >= N.  grid = Npad, block = 256.
// ---------------------------------------------------------------------------
__global__ __launch_bounds__(256)
void k_wprep(const float* __restrict__ src, unsigned short* __restrict__ dst, int N) {
    int j = blockIdx.x;              // output row (= weight column)
    int t = threadIdx.x;             // K index
    dst[(size_t)j * DMODEL + t] = (j < N) ? f2bf(src[(size_t)t * N + j])
                                          : (unsigned short)0;
}

// ---------------------------------------------------------------------------
// K1: embedding gather-sum.  grid = (4096, 4 mods), block = 256 (one d each).
// ---------------------------------------------------------------------------
__global__ __launch_bounds__(256)
void k_embed(const int* __restrict__ s0, const int* __restrict__ s1,
             const int* __restrict__ s2, const int* __restrict__ s3,
             const float* __restrict__ E0, const float* __restrict__ E1,
             const float* __restrict__ E2, const float* __restrict__ E3,
             float* __restrict__ Hf, unsigned short* __restrict__ Hb) {
    int r = blockIdx.x, m = blockIdx.y, d = threadIdx.x;
    const int*   sq = (m == 0) ? s0 : (m == 1) ? s1 : (m == 2) ? s2 : s3;
    const float* E  = (m == 0) ? E0 : (m == 1) ? E1 : (m == 2) ? E2 : E3;
    float acc = 0.f;
    #pragma unroll
    for (int c = 0; c < NCODES; ++c)
        acc += E[(size_t)sq[r * NCODES + c] * DMODEL + d];
    size_t o = ((size_t)m * N_TOK + r) * DMODEL + d;
    Hf[o] = acc;
    Hb[o] = f2bf(acc);
}

// ---------------------------------------------------------------------------
// K2: fused similarity GEMM (bf16 WMMA, f32 acc) + streaming top-16 + softmax.
//     One wave (32 threads) owns a 16-row stripe; streams 256 column tiles
//     with double-buffered B operands so tile jt+1's 16 b128 loads are in
//     flight while tile jt's WMMAs and top-k merge execute.
//     grid = (256 stripes, 4 mods), block = 32.
// ---------------------------------------------------------------------------
__global__ __launch_bounds__(32)
void k_simtopk(const unsigned short* __restrict__ Hb,
               int* __restrict__ tki, float* __restrict__ tkw) {
    const int stripe = blockIdx.x;
    const int mod    = blockIdx.y;
    const int lane   = threadIdx.x;
    const int lm     = lane & 15;
    const int hi     = lane >> 4;
    const int I      = stripe * 16;

    const unsigned short* H = Hb + (size_t)mod * N_TOK * DMODEL;

    __shared__ float smem[256];

    // Preload A stripe: 16 rows x 256 K, per ISA A layout (e<8 -> K=kb+hi*8+e,
    // e>=8 -> K=kb+16+hi*8+(e-8)).
    v16bf Areg[8];
    {
        const int rowA = I + lm;
        #pragma unroll
        for (int kc = 0; kc < 8; ++kc) {
            const unsigned short* p = H + (size_t)rowA * DMODEL + kc * 32 + hi * 8;
            Areg[kc] = load16bf(p, p + 16);
        }
    }

    // Per-row top-K state, sorted descending; lanes 0..15 own rows I+lane.
    float bv[KTOP];
    int   bi[KTOP];
    #pragma unroll
    for (int t = 0; t < KTOP; ++t) { bv[t] = -__builtin_inff(); bi[t] = 0; }

    // B layout: same as A (e<8 -> K=kc*32+hi*8+e, e>=8 -> K=kc*32+16+hi*8+(e-8)).
    auto loadB = [&](v16bf (&B)[8], int jt) {
        const int colRow = jt * 16 + lm;
        #pragma unroll
        for (int kc = 0; kc < 8; ++kc) {
            const unsigned short* p = H + (size_t)colRow * DMODEL + kc * 32 + hi * 8;
            B[kc] = load16bf(p, p + 16);
        }
    };

    auto processTile = [&](v16bf (&B)[8], int jt) {
        v8f acc = {};
        #pragma unroll
        for (int kc = 0; kc < 8; ++kc)
            acc = wmma_bf16(Areg[kc], B[kc], acc);
        // Mask diagonal (row == col) when this tile sits on it.
        if (jt == stripe) {
            #pragma unroll
            for (int v = 0; v < 8; ++v)
                if (v + 8 * hi == lm) acc[v] = -__builtin_inff();
        }
        // Stage tile to LDS: (M = v+8*hi, N = lm).
        #pragma unroll
        for (int v = 0; v < 8; ++v)
            smem[(v + 8 * hi) * 16 + lm] = acc[v];
        __syncthreads();

        if (lane < 16) {
            #pragma unroll 4
            for (int c = 0; c < 16; ++c) {
                float v = smem[lane * 16 + c];
                if (v > bv[KTOP - 1]) {            // threshold guard (common reject)
                    int j = jt * 16 + c;
                    int p = 0;
                    #pragma unroll
                    for (int t = 0; t < KTOP - 1; ++t) p += (bv[t] >= v) ? 1 : 0;
                    #pragma unroll
                    for (int t = KTOP - 1; t > 0; --t) {
                        bool sh = (t > p);
                        bv[t] = sh ? bv[t - 1] : bv[t];
                        bi[t] = sh ? bi[t - 1] : bi[t];
                    }
                    #pragma unroll
                    for (int t = 0; t < KTOP; ++t)
                        if (t == p) { bv[t] = v; bi[t] = j; }
                }
            }
        }
        __syncthreads();
    };

    const int NT = N_TOK / 16;
    v16bf B0[8], B1[8];
    loadB(B0, 0);
    for (int jt = 0; jt < NT; jt += 2) {
        loadB(B1, jt + 1);                 // prefetch odd tile
        processTile(B0, jt);
        const int nx = (jt + 2 < NT) ? jt + 2 : jt;   // harmless dummy on last iter
        loadB(B0, nx);                     // prefetch next even tile
        processTile(B1, jt + 1);
    }

    // Softmax over the 16 kept values (sorted desc -> bv[0] is the max).
    if (lane < 16) {
        const int row = I + lane;
        float w[KTOP], s = 0.f;
        #pragma unroll
        for (int t = 0; t < KTOP; ++t) { w[t] = __expf(bv[t] - bv[0]); s += w[t]; }
        float inv = 1.f / s;
        size_t base = ((size_t)mod * N_TOK + row) * KTOP;
        #pragma unroll
        for (int t = 0; t < KTOP; ++t) {
            tkw[base + t] = w[t] * inv;
            tki[base + t] = bi[t];
        }
    }
}

// ---------------------------------------------------------------------------
// K3: weighted gather + residual -> hbar (bf16).  grid = (4096, 4), block 256.
// ---------------------------------------------------------------------------
__global__ __launch_bounds__(256)
void k_attend(const float* __restrict__ Hf, const int* __restrict__ tki,
              const float* __restrict__ tkw, unsigned short* __restrict__ Hbar) {
    int r = blockIdx.x, m = blockIdx.y, d = threadIdx.x;
    const float* H  = Hf + (size_t)m * N_TOK * DMODEL;
    size_t kb = ((size_t)m * N_TOK + r) * KTOP;
    float acc = H[(size_t)r * DMODEL + d];
    #pragma unroll
    for (int t = 0; t < KTOP; ++t)
        acc += tkw[kb + t] * H[(size_t)tki[kb + t] * DMODEL + d];
    Hbar[((size_t)m * N_TOK + r) * DMODEL + d] = f2bf(acc);
}

// ---------------------------------------------------------------------------
// K4: generic WMMA GEMM.  One wave computes a 16x64 strip:
//     A tile (16x256 bf16) loaded once into 64 VGPRs, then 4 J-sub-tiles,
//     each with all 16 B loads issued before the 8-WMMA chain so stores of
//     sub-tile s overlap loads of sub-tile s+1.
//     C[M x N] = A[M x 256] x Bt[Npad x 256](col-major) + bias, opt ReLU,
//     optional f32 / bf16 outputs.  grid = (M/16, Npad/64), block = 32.
// ---------------------------------------------------------------------------
__global__ __launch_bounds__(32)
void k_gemm(const unsigned short* __restrict__ A,
            const unsigned short* __restrict__ Bt,
            const float* __restrict__ bias,
            float* __restrict__ Cf, unsigned short* __restrict__ Cb,
            int N, int ldcf, int ldcb, int relu) {
    const int lane = threadIdx.x, lm = lane & 15, hi = lane >> 4;
    const int I  = blockIdx.x * 16;
    const int J0 = blockIdx.y * 64;
    const int rowA = I + lm;

    v16bf Areg[8];
    #pragma unroll
    for (int kc = 0; kc < 8; ++kc) {
        const unsigned short* pa = A + (size_t)rowA * DMODEL + kc * 32 + hi * 8;
        Areg[kc] = load16bf(pa, pa + 16);
    }

    #pragma unroll
    for (int s = 0; s < 4; ++s) {
        const int J = J0 + s * 16;
        const int colB = J + lm;

        v16bf Breg[8];
        #pragma unroll
        for (int kc = 0; kc < 8; ++kc) {
            const unsigned short* pb = Bt + (size_t)colB * DMODEL + kc * 32 + hi * 8;
            Breg[kc] = load16bf(pb, pb + 16);
        }

        v8f acc = {};
        #pragma unroll
        for (int kc = 0; kc < 8; ++kc)
            acc = wmma_bf16(Areg[kc], Breg[kc], acc);

        const int j = J + lm;
        const float bb = (j < N) ? bias[j] : 0.f;
        #pragma unroll
        for (int v = 0; v < 8; ++v) {
            float x = acc[v] + bb;
            if (relu) x = fmaxf(x, 0.f);
            int r = I + v + 8 * hi;
            if (j < N) {
                if (Cf) Cf[(size_t)r * ldcf + j] = x;
                if (Cb) Cb[(size_t)r * ldcb + j] = f2bf(x);
            }
        }
    }
}

// ---------------------------------------------------------------------------
// K5: reparameterize + scatter raw mu/logvar into concat outputs.
//     grid = (4096, 4), block = 256.
// ---------------------------------------------------------------------------
__global__ __launch_bounds__(256)
void k_reparam(const float* __restrict__ enc_out,   // [4][4096][512]
               const float* __restrict__ e0, const float* __restrict__ e1,
               const float* __restrict__ e2, const float* __restrict__ e3,
               unsigned short* __restrict__ Zb,
               float* __restrict__ miu, float* __restrict__ logvar) {
    int r = blockIdx.x, m = blockIdx.y, d = threadIdx.x;
    size_t base  = ((size_t)m * N_TOK + r) * 512;
    size_t base0 = (size_t)r * 512;
    float mu  = enc_out[base + d];
    float lv  = enc_out[base + 256 + d];
    float dmu = enc_out[base0 + d];
    float dlv = enc_out[base0 + 256 + d];
    float mu2 = (m == 0) ? mu : mu + dmu;
    float lv2 = (m == 0) ? lv : lv + dlv;
    const float* eps = (m == 0) ? e0 : (m == 1) ? e1 : (m == 2) ? e2 : e3;
    float z = mu2 + eps[(size_t)r * DMODEL + d] * __expf(0.5f * lv2);
    Zb[((size_t)m * N_TOK + r) * DMODEL + d] = f2bf(z);
    miu   [(size_t)r * 1024 + m * DMODEL + d] = mu;
    logvar[(size_t)r * 1024 + m * DMODEL + d] = lv;
}

// ---------------------------------------------------------------------------
// Host launcher.
// ---------------------------------------------------------------------------
extern "C" void kernel_launch(void* const* d_in, const int* in_sizes, int n_in,
                              void* d_out, int out_size, void* d_ws, size_t ws_size,
                              hipStream_t stream) {
    (void)in_sizes; (void)n_in; (void)out_size; (void)ws_size;

    // Inputs (setup_inputs order)
    const int*   seq[4]  = {(const int*)d_in[0], (const int*)d_in[1],
                            (const int*)d_in[2], (const int*)d_in[3]};
    const float* eps[4]  = {(const float*)d_in[4], (const float*)d_in[5],
                            (const float*)d_in[6], (const float*)d_in[7]};
    const float* Emb[4]  = {(const float*)d_in[8], (const float*)d_in[9],
                            (const float*)d_in[10], (const float*)d_in[11]};
    const float* enc_W = (const float*)d_in[12];
    const float* enc_b = (const float*)d_in[13];
    const float* dec_W = (const float*)d_in[14];
    const float* dec_b = (const float*)d_in[15];
    const float *W1[4], *b1[4], *W2[4], *b2[4];
    for (int m = 0; m < 4; ++m) {
        W1[m] = (const float*)d_in[16 + 4 * m];
        b1[m] = (const float*)d_in[17 + 4 * m];
        W2[m] = (const float*)d_in[18 + 4 * m];
        b2[m] = (const float*)d_in[19 + 4 * m];
    }

    const int Vocab[4] = {2000, 500, 1000, 800};
    const int Vpad[4]  = {2048, 512, 1024, 832};   // multiples of 64

    // Workspace carve-up
    char* p = (char*)d_ws;
    auto carve = [&](size_t bytes) {
        void* q = (void*)p;
        p += (bytes + 255) & ~(size_t)255;
        return q;
    };
    const size_t ND = (size_t)N_TOK * DMODEL;
    float*          Hf      = (float*)         carve(4 * ND * 4);       // h (f32)
    unsigned short* HbW     = (unsigned short*)carve(4 * ND * 2);       // h (bf16)
    int*            tki     = (int*)           carve(4 * (size_t)N_TOK * KTOP * 4);
    float*          tkw     = (float*)         carve(4 * (size_t)N_TOK * KTOP * 4);
    unsigned short* Hbar    = (unsigned short*)carve(4 * ND * 2);       // hbar bf16
    float*          enc_out = (float*)         carve(4 * (size_t)N_TOK * 512 * 4);
    unsigned short* Zb      = (unsigned short*)carve(4 * ND * 2);
    unsigned short* Vgenb   = (unsigned short*)carve(4 * ND * 2);
    unsigned short* Hidb    = (unsigned short*)carve(4 * ND * 2);
    unsigned short* encWt   = (unsigned short*)carve((size_t)512 * 256 * 2);
    unsigned short* decWt   = (unsigned short*)carve((size_t)256 * 256 * 2);
    unsigned short* W1t[4];
    for (int m = 0; m < 4; ++m) W1t[m] = (unsigned short*)carve((size_t)256 * 256 * 2);
    unsigned short* W2t[4];
    for (int m = 0; m < 4; ++m) W2t[m] = (unsigned short*)carve((size_t)Vpad[m] * 256 * 2);

    // Output layout (flat, return order)
    float* out = (float*)d_out;
    size_t o = 0;
    float* out_logits[4];
    for (int m = 0; m < 4; ++m) { out_logits[m] = out + o; o += (size_t)N_TOK * Vocab[m]; }
    float* out_miu = out + o;  o += (size_t)N_TOK * 1024;
    float* out_lv  = out + o;

    // --- K0: weight prep (bf16, col-major, padded) ---
    k_wprep<<<dim3(512), dim3(256), 0, stream>>>(enc_W, encWt, 512);
    k_wprep<<<dim3(256), dim3(256), 0, stream>>>(dec_W, decWt, 256);
    for (int m = 0; m < 4; ++m)
        k_wprep<<<dim3(256), dim3(256), 0, stream>>>(W1[m], W1t[m], 256);
    for (int m = 0; m < 4; ++m)
        k_wprep<<<dim3(Vpad[m]), dim3(256), 0, stream>>>(W2[m], W2t[m], Vocab[m]);

    // --- K1: embedding sums ---
    k_embed<<<dim3(N_TOK, 4), dim3(256), 0, stream>>>(
        seq[0], seq[1], seq[2], seq[3], Emb[0], Emb[1], Emb[2], Emb[3], Hf, HbW);

    // --- K2: fused sim GEMM + top-16 + softmax ---
    k_simtopk<<<dim3(N_TOK / 16, 4), dim3(32), 0, stream>>>(HbW, tki, tkw);

    // --- K3: weighted gather + residual ---
    k_attend<<<dim3(N_TOK, 4), dim3(256), 0, stream>>>(Hf, tki, tkw, Hbar);

    // --- K4a: encoder GEMMs (N=512) ---
    for (int m = 0; m < 4; ++m)
        k_gemm<<<dim3(N_TOK / 16, 512 / 64), dim3(32), 0, stream>>>(
            Hbar + (size_t)m * ND, encWt, enc_b,
            enc_out + (size_t)m * N_TOK * 512, nullptr, 512, 512, 0, 0);

    // --- K5: reparameterize + concat mu/logvar outputs ---
    k_reparam<<<dim3(N_TOK, 4), dim3(256), 0, stream>>>(
        enc_out, eps[0], eps[1], eps[2], eps[3], Zb, out_miu, out_lv);

    // --- K4b: decoder GEMMs (N=256) -> vgen (bf16 only) ---
    for (int m = 0; m < 4; ++m)
        k_gemm<<<dim3(N_TOK / 16, 256 / 64), dim3(32), 0, stream>>>(
            Zb + (size_t)m * ND, decWt, dec_b,
            nullptr, Vgenb + (size_t)m * ND, 256, 0, 256, 0);

    // --- K4c: MLP layer 1 (ReLU) ---
    for (int m = 0; m < 4; ++m)
        k_gemm<<<dim3(N_TOK / 16, 256 / 64), dim3(32), 0, stream>>>(
            Vgenb + (size_t)m * ND, W1t[m], b1[m],
            nullptr, Hidb + (size_t)m * ND, 256, 0, 256, 1);

    // --- K4d: MLP layer 2 -> logits (f32, bounds-checked on vocab) ---
    for (int m = 0; m < 4; ++m)
        k_gemm<<<dim3(N_TOK / 16, Vpad[m] / 64), dim3(32), 0, stream>>>(
            Hidb + (size_t)m * ND, W2t[m], b2[m],
            out_logits[m], nullptr, Vocab[m], Vocab[m], 0, 0);
}
